// MultiHeadAttention_10282151706682
// MI455X (gfx1250) — matrix operand order repaired and checked by reading, not compiled, not run on hardware
//
#include <hip/hip_runtime.h>
#include <hip/hip_bf16.h>
#include <stdint.h>

typedef __attribute__((ext_vector_type(16))) _Float16 v16h;
typedef __attribute__((ext_vector_type(8)))  _Float16 v8h;
typedef __attribute__((ext_vector_type(8)))  float    v8f;

#define BQ  2
#define SQ  2048
#define DM  1024
#define NH  16
#define HD2 64

// ---------------------------------------------------------------------------
// Async global->LDS copy of 16 bytes (CDNA5, ASYNCcnt-tracked).
// ldsoff: wave-relative LDS byte address (low 32 bits of flat LDS address).
// ---------------------------------------------------------------------------
__device__ __forceinline__ void async_copy_b128(unsigned ldsoff,
                                                const void* gptr) {
  unsigned long long ga = (unsigned long long)(uintptr_t)gptr;
  asm volatile("global_load_async_to_lds_b128 %0, %1, off"
               :: "v"(ldsoff), "v"(ga) : "memory");
}
__device__ __forceinline__ void wait_asynccnt0() {
  asm volatile("s_wait_asynccnt 0x0" ::: "memory");
}
__device__ __forceinline__ unsigned lds_addr_of(const void* p) {
  return (unsigned)(uintptr_t)p;   // flat LDS addr: low 32 bits = LDS offset
}

// ---------------------------------------------------------------------------
// WMMA fragment helpers (layouts per CDNA5 ISA 7.12.2, wave32)
// ---------------------------------------------------------------------------
__device__ __forceinline__ v16h load_A_frag(const _Float16* a, int lda) {
  const int lane = threadIdx.x & 31;
  const int m  = lane & 15;
  const int kb = (lane >> 4) << 3;       // 0 or 8
  const _Float16* p = a + m * lda + kb;
  v8h lo = *(const v8h*)(p);
  v8h hi = *(const v8h*)(p + 16);
  v16h r;
#pragma unroll
  for (int i = 0; i < 8; ++i) { r[i] = lo[i]; r[i + 8] = hi[i]; }
  return r;
}

__device__ __forceinline__ v16h load_B_frag(const _Float16* bt, int ldb) {
  const int lane = threadIdx.x & 31;
  const int n  = lane & 15;
  const int kb = (lane >> 4) << 3;       // 0 or 8
  const _Float16* p = bt + n * ldb + kb;
  v8h lo = *(const v8h*)(p);
  v8h hi = *(const v8h*)(p + 16);
  v16h r;
#pragma unroll
  for (int i = 0; i < 8; ++i) { r[i] = lo[i]; r[i + 8] = hi[i]; }
  return r;
}

__device__ __forceinline__ v8f wmma16(v16h a, v16h b, v8f c) {
  return __builtin_amdgcn_wmma_f32_16x16x32_f16(
      false, a, false, b, (short)0, c, false, false);
}

// ---------------------------------------------------------------------------
// Kernel 1: fused Q/K/V projection.  Y = X @ W + b, written head-split f16.
// ---------------------------------------------------------------------------
__global__ __launch_bounds__(256) void proj_kernel(
    const float* __restrict__ qin, const float* __restrict__ kin,
    const float* __restrict__ vin,
    const float* __restrict__ wq, const float* __restrict__ bq,
    const float* __restrict__ wk, const float* __restrict__ bk,
    const float* __restrict__ wv, const float* __restrict__ bv,
    _Float16* __restrict__ Qh, _Float16* __restrict__ Kh,
    _Float16* __restrict__ Vh)
{
  __shared__ _Float16 lds_a[128][48];
  __shared__ _Float16 lds_bt[64][48];

  const float* Xin; const float* W; const float* bias; _Float16* dst;
  if (blockIdx.z == 0)      { Xin = qin; W = wq; bias = bq; dst = Qh; }
  else if (blockIdx.z == 1) { Xin = kin; W = wk; bias = bk; dst = Kh; }
  else                      { Xin = vin; W = wv; bias = bv; dst = Vh; }

  const int m0   = blockIdx.x * 128;
  const int n0   = blockIdx.y * 64;
  const int t    = threadIdx.x;
  const int w    = t >> 5;
  const int lane = t & 31;
  const int wm   = w >> 1;
  const int wn   = w & 1;
  const int ln   = lane & 15, hl = lane >> 4;

  v8f acc[2][2];
#pragma unroll
  for (int i = 0; i < 2; ++i)
#pragma unroll
    for (int j = 0; j < 2; ++j) acc[i][j] = (v8f){};

  const int arow = t >> 1;
  const int ach  = (t & 1) * 16;
  const int brow = t >> 3;
  const int bnc  = (t & 7) * 8;

  for (int k0 = 0; k0 < DM; k0 += 32) {
    { // A tile: 128x32 fp32 -> f16 (conversion forces VGPR path)
      const float* src = Xin + (size_t)(m0 + arow) * DM + k0 + ach;
      float4 f0 = *(const float4*)(src + 0);
      float4 f1 = *(const float4*)(src + 4);
      float4 f2 = *(const float4*)(src + 8);
      float4 f3 = *(const float4*)(src + 12);
      v8h h0, h1;
      h0[0] = (_Float16)f0.x; h0[1] = (_Float16)f0.y;
      h0[2] = (_Float16)f0.z; h0[3] = (_Float16)f0.w;
      h0[4] = (_Float16)f1.x; h0[5] = (_Float16)f1.y;
      h0[6] = (_Float16)f1.z; h0[7] = (_Float16)f1.w;
      h1[0] = (_Float16)f2.x; h1[1] = (_Float16)f2.y;
      h1[2] = (_Float16)f2.z; h1[3] = (_Float16)f2.w;
      h1[4] = (_Float16)f3.x; h1[5] = (_Float16)f3.y;
      h1[6] = (_Float16)f3.z; h1[7] = (_Float16)f3.w;
      *(v8h*)&lds_a[arow][ach]     = h0;
      *(v8h*)&lds_a[arow][ach + 8] = h1;
    }
    { // B tile: 32x64 fp32, transposed into (n,k)
      const float* src = W + (size_t)(k0 + brow) * DM + n0 + bnc;
      float4 f0 = *(const float4*)(src + 0);
      float4 f1 = *(const float4*)(src + 4);
      lds_bt[bnc + 0][brow] = (_Float16)f0.x;
      lds_bt[bnc + 1][brow] = (_Float16)f0.y;
      lds_bt[bnc + 2][brow] = (_Float16)f0.z;
      lds_bt[bnc + 3][brow] = (_Float16)f0.w;
      lds_bt[bnc + 4][brow] = (_Float16)f1.x;
      lds_bt[bnc + 5][brow] = (_Float16)f1.y;
      lds_bt[bnc + 6][brow] = (_Float16)f1.z;
      lds_bt[bnc + 7][brow] = (_Float16)f1.w;
    }
    __syncthreads();

    v16h a0 = load_A_frag(&lds_a[wm * 32 +  0][0], 48);
    v16h a1 = load_A_frag(&lds_a[wm * 32 + 16][0], 48);
    v16h b0 = load_B_frag(&lds_bt[wn * 32 +  0][0], 48);
    v16h b1 = load_B_frag(&lds_bt[wn * 32 + 16][0], 48);
    acc[0][0] = wmma16(a0, b0, acc[0][0]);
    acc[0][1] = wmma16(a0, b1, acc[0][1]);
    acc[1][0] = wmma16(a1, b0, acc[1][0]);
    acc[1][1] = wmma16(a1, b1, acc[1][1]);
    __syncthreads();
  }

#pragma unroll
  for (int i = 0; i < 2; ++i) {
#pragma unroll
    for (int j = 0; j < 2; ++j) {
      const int n  = n0 + wn * 32 + j * 16 + ln;
      const float bn = bias[n];
      const int h = n >> 6, d = n & 63;
#pragma unroll
      for (int v = 0; v < 8; ++v) {
        const int m    = m0 + wm * 32 + i * 16 + v + 8 * hl;
        const int bidx = m >> 11;
        const int s    = m & 2047;
        dst[((size_t)(bidx * NH + h) * SQ + s) * HD2 + d] =
            (_Float16)(acc[i][j][v] + bn);
      }
    }
  }
}

// ---------------------------------------------------------------------------
// Kernel 2: flash attention with double-buffered async K staging.
// Block = 128 queries (8 waves x 16 rows); 64-key chunks; online softmax.
// ---------------------------------------------------------------------------
__global__ __launch_bounds__(256) void attn_kernel(
    const _Float16* __restrict__ Qh, const _Float16* __restrict__ Kh,
    const _Float16* __restrict__ Vh, const unsigned char* __restrict__ mask,
    _Float16* __restrict__ CTX)
{
  __shared__ _Float16 lds_k [2][64][72];  // K chunk, row-major (= B layout)
  __shared__ _Float16 lds_vt[2][64][72];  // V chunk transposed: (d, key)
  __shared__ _Float16 lds_p [8][16][72];  // per-wave P bounce

  const int qt = blockIdx.x, hh = blockIdx.y, bb = blockIdx.z;
  const int t = threadIdx.x, w = t >> 5, lane = t & 31;
  const int ln = lane & 15, hl = lane >> 4;

  const size_t headoff = (size_t)(bb * NH + hh) * SQ * HD2;
  const int q0 = qt * 128 + w * 16;

  const v16h aq0 = load_A_frag(Qh + headoff + (size_t)q0 * HD2,      HD2);
  const v16h aq1 = load_A_frag(Qh + headoff + (size_t)q0 * HD2 + 32, HD2);

  v8f o[4];
#pragma unroll
  for (int jd = 0; jd < 4; ++jd) o[jd] = (v8f){};
  float mstat[8], lstat[8];
#pragma unroll
  for (int v = 0; v < 8; ++v) { mstat[v] = -3.0e38f; lstat[v] = 0.0f; }

  const int ckey = t >> 2;                // 0..63
  const int cdb  = (t & 3) * 16;          // 0,16,32,48
  const float sc = 1.0f / (8.0f + 1e-8f);
  const unsigned char* mbase = mask + (size_t)bb * SQ * SQ;

  v8h vy0, vy1;   // staged V registers (transpose-scatter deferred)

  // issue: K chunk via async-to-LDS; V chunk into registers
  auto stage_issue = [&](int kc, int buf) {
    const _Float16* ks = Kh + headoff + (size_t)(kc + ckey) * HD2 + cdb;
    const unsigned kd = lds_addr_of(&lds_k[buf][ckey][cdb]);
    async_copy_b128(kd,      ks);
    async_copy_b128(kd + 16, ks + 8);
    const _Float16* vs = Vh + headoff + (size_t)(kc + ckey) * HD2 + cdb;
    vy0 = *(const v8h*)(vs);
    vy1 = *(const v8h*)(vs + 8);
  };
  // commit: transpose-scatter staged V registers into LDS
  auto stage_commit = [&](int buf) {
#pragma unroll
    for (int i = 0; i < 8; ++i) lds_vt[buf][cdb + i][ckey]     = vy0[i];
#pragma unroll
    for (int i = 0; i < 8; ++i) lds_vt[buf][cdb + 8 + i][ckey] = vy1[i];
  };

  stage_issue(0, 0);
  stage_commit(0);
  wait_asynccnt0();
  __syncthreads();

  for (int kc = 0; kc < SQ; kc += 64) {
    const int cur = (kc >> 6) & 1;
    const bool pre = (kc + 64) < SQ;
    if (pre) stage_issue(kc + 64, cur ^ 1);   // overlaps with compute below

    // S = Q K^T
    v8f s[4];
#pragma unroll
    for (int j = 0; j < 4; ++j) {
      v16h bk0 = load_B_frag(&lds_k[cur][j * 16][0],      72);
      v16h bk1 = load_B_frag(&lds_k[cur][j * 16][0] + 32, 72);
      s[j] = wmma16(aq0, bk0, (v8f){});
      s[j] = wmma16(aq1, bk1, s[j]);
    }

    // scale + mask + per-row max (C/D layout: row M = v + 8*hl, col N = ln)
    float mx[8];
#pragma unroll
    for (int v = 0; v < 8; ++v) mx[v] = -3.0e38f;
#pragma unroll
    for (int j = 0; j < 4; ++j) {
      const int col = kc + j * 16 + ln;
#pragma unroll
      for (int v = 0; v < 8; ++v) {
        const int qr = q0 + v + 8 * hl;
        float val = s[j][v] * sc;
        val = mbase[(size_t)qr * SQ + col] ? -3.0e38f : val;
        s[j][v] = val;
        mx[v] = fmaxf(mx[v], val);
      }
    }
#pragma unroll
    for (int off = 1; off < 16; off <<= 1)
#pragma unroll
      for (int v = 0; v < 8; ++v)
        mx[v] = fmaxf(mx[v], __shfl_xor(mx[v], off, 32));

    float mnew[8], corr[8], rs[8];
#pragma unroll
    for (int v = 0; v < 8; ++v) {
      mnew[v] = fmaxf(mstat[v], mx[v]);
      corr[v] = __expf(mstat[v] - mnew[v]);
      rs[v]   = 0.0f;
    }
#pragma unroll
    for (int j = 0; j < 4; ++j)
#pragma unroll
      for (int v = 0; v < 8; ++v) {
        float p = __expf(s[j][v] - mnew[v]);
        s[j][v] = p;
        rs[v] += p;
      }
#pragma unroll
    for (int off = 1; off < 16; off <<= 1)
#pragma unroll
      for (int v = 0; v < 8; ++v) rs[v] += __shfl_xor(rs[v], off, 32);
#pragma unroll
    for (int v = 0; v < 8; ++v) {
      lstat[v] = lstat[v] * corr[v] + rs[v];
      mstat[v] = mnew[v];
    }
#pragma unroll
    for (int jd = 0; jd < 4; ++jd)
#pragma unroll
      for (int v = 0; v < 8; ++v) o[jd][v] *= corr[v];

    // P: C/D layout -> A layout via per-wave LDS bounce
#pragma unroll
    for (int j = 0; j < 4; ++j)
#pragma unroll
      for (int v = 0; v < 8; ++v)
        lds_p[w][v + 8 * hl][j * 16 + ln] = (_Float16)s[j][v];
    asm volatile("s_wait_dscnt 0x0" ::: "memory");   // wave-local LDS RAW

    v16h ap0 = load_A_frag(&lds_p[w][0][0],      72);
    v16h ap1 = load_A_frag(&lds_p[w][0][0] + 32, 72);
#pragma unroll
    for (int jd = 0; jd < 4; ++jd) {
      v16h bv0 = load_B_frag(&lds_vt[cur][jd * 16][0],      72);
      v16h bv1 = load_B_frag(&lds_vt[cur][jd * 16][0] + 32, 72);
      o[jd] = wmma16(ap0, bv0, o[jd]);
      o[jd] = wmma16(ap1, bv1, o[jd]);
    }

    if (pre) stage_commit(cur ^ 1);   // V global latency hidden by WMMAs
    wait_asynccnt0();                 // next K chunk landed in LDS
    __syncthreads();
  }

  // ctx /= l ; store f16 into [B*S, H*64]
#pragma unroll
  for (int jd = 0; jd < 4; ++jd)
#pragma unroll
    for (int v = 0; v < 8; ++v) {
      const int qr = q0 + v + 8 * hl;
      float val = o[jd][v] / lstat[v];
      CTX[((size_t)(bb * SQ + qr)) * DM + hh * HD2 + jd * 16 + ln] =
          (_Float16)val;
    }
}

// ---------------------------------------------------------------------------
// Kernel 3: output projection + bias + residual -> fp32 pre-LN buffer.
// A tile is already f16 -> pure async global->LDS copy (no VGPR round trip).
// ---------------------------------------------------------------------------
__global__ __launch_bounds__(256) void outproj_kernel(
    const _Float16* __restrict__ CTX, const float* __restrict__ wo,
    const float* __restrict__ bo, const float* __restrict__ resid,
    float* __restrict__ Xout)
{
  __shared__ _Float16 lds_a[128][48];
  __shared__ _Float16 lds_bt[64][48];

  const int m0 = blockIdx.x * 128;
  const int n0 = blockIdx.y * 64;
  const int t = threadIdx.x, w = t >> 5, lane = t & 31;
  const int wm = w >> 1, wn = w & 1;
  const int ln = lane & 15, hl = lane >> 4;

  v8f acc[2][2];
#pragma unroll
  for (int i = 0; i < 2; ++i)
#pragma unroll
    for (int j = 0; j < 2; ++j) acc[i][j] = (v8f){};

  const int arow = t >> 1;
  const int ach  = (t & 1) * 16;
  const int brow = t >> 3;
  const int bnc  = (t & 7) * 8;

  for (int k0 = 0; k0 < DM; k0 += 32) {
    { // A tile: f16 copy via async-to-LDS
      const _Float16* src = CTX + (size_t)(m0 + arow) * DM + k0 + ach;
      const unsigned ad = lds_addr_of(&lds_a[arow][ach]);
      async_copy_b128(ad,      src);
      async_copy_b128(ad + 16, src + 8);
    }
    { // B tile: 32x64 fp32, transposed into (n,k)
      const float* src = wo + (size_t)(k0 + brow) * DM + n0 + bnc;
      float4 f0 = *(const float4*)(src + 0);
      float4 f1 = *(const float4*)(src + 4);
      lds_bt[bnc + 0][brow] = (_Float16)f0.x;
      lds_bt[bnc + 1][brow] = (_Float16)f0.y;
      lds_bt[bnc + 2][brow] = (_Float16)f0.z;
      lds_bt[bnc + 3][brow] = (_Float16)f0.w;
      lds_bt[bnc + 4][brow] = (_Float16)f1.x;
      lds_bt[bnc + 5][brow] = (_Float16)f1.y;
      lds_bt[bnc + 6][brow] = (_Float16)f1.z;
      lds_bt[bnc + 7][brow] = (_Float16)f1.w;
    }
    wait_asynccnt0();
    __syncthreads();

    v16h a0 = load_A_frag(&lds_a[wm * 32 +  0][0], 48);
    v16h a1 = load_A_frag(&lds_a[wm * 32 + 16][0], 48);
    v16h b0 = load_B_frag(&lds_bt[wn * 32 +  0][0], 48);
    v16h b1 = load_B_frag(&lds_bt[wn * 32 + 16][0], 48);
    acc[0][0] = wmma16(a0, b0, acc[0][0]);
    acc[0][1] = wmma16(a0, b1, acc[0][1]);
    acc[1][0] = wmma16(a1, b0, acc[1][0]);
    acc[1][1] = wmma16(a1, b1, acc[1][1]);
    __syncthreads();
  }

#pragma unroll
  for (int i = 0; i < 2; ++i)
#pragma unroll
    for (int j = 0; j < 2; ++j) {
      const int n = n0 + wn * 32 + j * 16 + ln;
      const float bn = bo[n];
#pragma unroll
      for (int v = 0; v < 8; ++v) {
        const int m = m0 + wm * 32 + i * 16 + v + 8 * hl;
        Xout[(size_t)m * DM + n] =
            acc[i][j][v] + bn + resid[(size_t)m * DM + n];
      }
    }
}

// ---------------------------------------------------------------------------
// Kernel 4: LayerNorm over the last dim (1024), one block per row
// ---------------------------------------------------------------------------
__global__ __launch_bounds__(256) void layernorm_kernel(
    const float* __restrict__ X, const float* __restrict__ gamma,
    const float* __restrict__ beta, float* __restrict__ out)
{
  const int r = blockIdx.x;
  const int t = threadIdx.x;
  const int c0 = t * 4;
  const float4 x4 = *(const float4*)(X + (size_t)r * DM + c0);
  float s1 = x4.x + x4.y + x4.z + x4.w;
  float s2 = x4.x * x4.x + x4.y * x4.y + x4.z * x4.z + x4.w * x4.w;
#pragma unroll
  for (int off = 16; off >= 1; off >>= 1) {
    s1 += __shfl_xor(s1, off, 32);
    s2 += __shfl_xor(s2, off, 32);
  }
  __shared__ float r1[8], r2[8];
  __shared__ float mu_s, rs_s;
  if ((t & 31) == 0) { r1[t >> 5] = s1; r2[t >> 5] = s2; }
  __syncthreads();
  if (t == 0) {
    float a = 0.0f, c = 0.0f;
#pragma unroll
    for (int i = 0; i < 8; ++i) { a += r1[i]; c += r2[i]; }
    const float mu  = a * (1.0f / DM);
    const float var = c * (1.0f / DM) - mu * mu;
    mu_s = mu;
    rs_s = rsqrtf(var + 1e-5f);
  }
  __syncthreads();
  const float mu = mu_s, rstd = rs_s;
  float4 o4;
  o4.x = (x4.x - mu) * rstd * gamma[c0 + 0] + beta[c0 + 0];
  o4.y = (x4.y - mu) * rstd * gamma[c0 + 1] + beta[c0 + 1];
  o4.z = (x4.z - mu) * rstd * gamma[c0 + 2] + beta[c0 + 2];
  o4.w = (x4.w - mu) * rstd * gamma[c0 + 3] + beta[c0 + 3];
  *(float4*)(out + (size_t)r * DM + c0) = o4;
}

// ---------------------------------------------------------------------------
extern "C" void kernel_launch(void* const* d_in, const int* in_sizes, int n_in,
                              void* d_out, int out_size, void* d_ws,
                              size_t ws_size, hipStream_t stream) {
  (void)in_sizes; (void)n_in; (void)out_size; (void)ws_size;
  const float* q  = (const float*)d_in[0];
  const float* k  = (const float*)d_in[1];
  const float* v  = (const float*)d_in[2];
  const unsigned char* mask = (const unsigned char*)d_in[3];
  const float* wq = (const float*)d_in[4];
  const float* bq = (const float*)d_in[5];
  const float* wk = (const float*)d_in[6];
  const float* bk = (const float*)d_in[7];
  const float* wv = (const float*)d_in[8];
  const float* bv = (const float*)d_in[9];
  const float* wo = (const float*)d_in[10];
  const float* bo = (const float*)d_in[11];
  const float* g  = (const float*)d_in[12];
  const float* be = (const float*)d_in[13];
  float* out = (float*)d_out;

  const size_t HE = (size_t)BQ * NH * SQ * HD2;   // 4,194,304 halfs (8 MB)
  _Float16* Qh  = (_Float16*)d_ws;
  _Float16* Kh  = Qh + HE;
  _Float16* Vh  = Kh + HE;
  _Float16* CTX = Vh + HE;
  float*    X   = (float*)(CTX + HE);             // 16 MB fp32 pre-LN

  dim3 blk(256);
  proj_kernel<<<dim3(32, 16, 3), blk, 0, stream>>>(
      q, k, v, wq, bq, wk, bk, wv, bv, Qh, Kh, Vh);
  attn_kernel<<<dim3(SQ / 128, NH, BQ), blk, 0, stream>>>(
      Qh, Kh, Vh, mask, CTX);
  outproj_kernel<<<dim3(32, 16), blk, 0, stream>>>(CTX, wo, bo, q, X);
  layernorm_kernel<<<dim3(BQ * SQ), blk, 0, stream>>>(X, g, be, out);
}
